// LinearAttention_85899345920270
// MI455X (gfx1250) — matrix operand order repaired and checked by reading, not compiled, not run on hardware
//
#include <hip/hip_runtime.h>
#include <hip/hip_bf16.h>

// ---------------------------------------------------------------------------
// Linear attention (elu feature map), B=4 N=8192 H=8 D=M=64, fp32 in/out.
// Memory-bound (~320MB @ 23.3TB/s ~ 14us); contraction on v_wmma_f32_16x16x32_f16.
// Phase 1: KV[b,h][m][d] = sum_n V[n,m]*K'[n,d]  and Ksum[b,h][d] (WMMA + atomics)
// Phase 2: Out[n,m] = (Q'_n . KV[m,:]) * 1/(Q'_n . Ksum + eps)     (WMMA)
// ---------------------------------------------------------------------------

typedef _Float16 h8   __attribute__((ext_vector_type(8)));
typedef _Float16 v16h __attribute__((ext_vector_type(16)));
typedef float    v8f  __attribute__((ext_vector_type(8)));

#define BQ   4
#define NQ   8192
#define HQ   8
#define DQ   64
#define BH   (BQ * HQ)
#define EPSL 1e-6f

#define P1_WAVES 4
#define P1_STEPS 4
#define P1_ROWS  (P1_WAVES * 32 * P1_STEPS)  // 512 sequence rows per block
#define PITCH    40                          // halves; 80B row pitch -> bank-conflict-free frags

__device__ __forceinline__ float elup1(float x) {
    // elu(x)+1 == x>0 ? x+1 : exp(x)
    return x > 0.0f ? x + 1.0f : __expf(x);
}

union Frag {
    v16h v;
    h8   h[2];
    _Float16 e[16];
};

// ---------------------------------------------------------------------------
__global__ void la_zero(float* __restrict__ p, int n) {
    int i = blockIdx.x * blockDim.x + threadIdx.x;
    if (i < n) p[i] = 0.0f;
}

// ---------------------------------------------------------------------------
// Phase 1: grid(NQ/P1_ROWS, BH), 128 threads (4 waves).
// Each wave: P1_STEPS steps of 32 seq rows. Per step: stage V^T and K'^T tiles
// to LDS (f16), 16 WMMAs (4 m-tiles x 4 d-tiles, K=32). Ksum from staged tile.
// ---------------------------------------------------------------------------
__global__ __launch_bounds__(128)
void la_phase1(const float* __restrict__ Kin, const float* __restrict__ Vin,
               const float* __restrict__ maskIn,
               float* __restrict__ kvOut, float* __restrict__ ksumOut)
{
    __shared__ float     kvLds[64 * 64];                // block KV accumulator
    __shared__ float     ksumLds[64];
    __shared__ _Float16  stA[P1_WAVES][64 * PITCH];     // V^T  tile: [m][n]
    __shared__ _Float16  stB[P1_WAVES][64 * PITCH];     // K'^T tile: [d][n]

    const int tid  = threadIdx.x;
    const int wave = tid >> 5;
    const int lane = tid & 31;
    const int col  = lane & 15;   // 0..15
    const int half = lane >> 4;   // 0..1
    const int bh   = blockIdx.y;
    const int b    = bh >> 3;
    const int h    = bh & 7;

    #pragma unroll
    for (int i = 0; i < 32; ++i) kvLds[tid + i * 128] = 0.0f;
    if (tid < 64) ksumLds[tid] = 0.0f;
    __syncthreads();

    v8f acc[16] = {};   // [mi][di] 16x16 f32 tiles

    _Float16* sa = &stA[wave][0];
    _Float16* sb = &stB[wave][0];

    const long rowStride = (long)HQ * DQ;
    const long base      = ((long)b * NQ) * rowStride + (long)h * DQ;

    for (int s = 0; s < P1_STEPS; ++s) {
        const int n = blockIdx.x * P1_ROWS + (wave * P1_STEPS + s) * 32 + lane;

        // ---- stage: lane owns sequence row n; write transposed f16 tiles ----
        const float4* vr = (const float4*)(Vin + base + (long)n * rowStride);
        const float4* kr = (const float4*)(Kin + base + (long)n * rowStride);
        const float   mk = maskIn[b * NQ + n];
        #pragma unroll
        for (int j = 0; j < 16; ++j) {
            float4 v4 = vr[j];
            sa[(j * 4 + 0) * PITCH + lane] = (_Float16)v4.x;
            sa[(j * 4 + 1) * PITCH + lane] = (_Float16)v4.y;
            sa[(j * 4 + 2) * PITCH + lane] = (_Float16)v4.z;
            sa[(j * 4 + 3) * PITCH + lane] = (_Float16)v4.w;
            float4 k4 = kr[j];
            sb[(j * 4 + 0) * PITCH + lane] = (_Float16)(elup1(k4.x) * mk);
            sb[(j * 4 + 1) * PITCH + lane] = (_Float16)(elup1(k4.y) * mk);
            sb[(j * 4 + 2) * PITCH + lane] = (_Float16)(elup1(k4.z) * mk);
            sb[(j * 4 + 3) * PITCH + lane] = (_Float16)(elup1(k4.w) * mk);
        }
        // same-wave LDS write -> read: drain DS counter (CDNA5 split counters)
        asm volatile("s_wait_dscnt 0" ::: "memory");

        // ---- Ksum partial: lane sums d-rows {lane, lane+32} over 32 n ----
        float s0 = 0.0f, s1 = 0.0f;
        #pragma unroll
        for (int j = 0; j < 4; ++j) {
            h8 x0 = *(const h8*)&sb[lane * PITCH + j * 8];
            h8 x1 = *(const h8*)&sb[(lane + 32) * PITCH + j * 8];
            #pragma unroll
            for (int e = 0; e < 8; ++e) { s0 += (float)x0[e]; s1 += (float)x1[e]; }
        }
        atomicAdd(&ksumLds[lane],      s0);
        atomicAdd(&ksumLds[lane + 32], s1);

        // ---- B fragments (K' tile, 32x16): lane=col d, k contiguous ----
        Frag bf[4];
        #pragma unroll
        for (int di = 0; di < 4; ++di) {
            const int d = di * 16 + col;
            bf[di].h[0] = *(const h8*)&sb[d * PITCH + half * 16];
            bf[di].h[1] = *(const h8*)&sb[d * PITCH + half * 16 + 8];
        }
        // ---- A fragments (V^T tile, 16x32) + 16 WMMAs ----
        #pragma unroll
        for (int mi = 0; mi < 4; ++mi) {
            Frag af;
            const int mrow = mi * 16 + col;
            af.h[0] = *(const h8*)&sa[mrow * PITCH + half * 16];       // K 0..7 / 16..23
            af.h[1] = *(const h8*)&sa[mrow * PITCH + half * 16 + 8];   // K 8..15 / 24..31
            #pragma unroll
            for (int di = 0; di < 4; ++di) {
                acc[mi * 4 + di] = __builtin_amdgcn_wmma_f32_16x16x32_f16(
                    false, af.v, false, bf[di].v, (short)0, acc[mi * 4 + di],
                    false, false);
            }
        }
        asm volatile("" ::: "memory");  // keep next step's stores below these loads
    }

    // ---- reduce wave tiles into block LDS accumulator ----
    #pragma unroll
    for (int mi = 0; mi < 4; ++mi)
        #pragma unroll
        for (int di = 0; di < 4; ++di)
            #pragma unroll
            for (int r = 0; r < 8; ++r) {
                const int M = mi * 16 + 8 * half + r;  // m index (C/D row layout)
                const int d = di * 16 + col;           // d index (C/D col layout)
                atomicAdd(&kvLds[M * 64 + d], acc[mi * 4 + di][r]);
            }
    __syncthreads();

    // ---- block -> global partial sums ----
    float* kvg = kvOut + bh * 4096;
    #pragma unroll
    for (int i = 0; i < 32; ++i)
        atomicAdd(&kvg[tid + i * 128], kvLds[tid + i * 128]);
    if (tid < 64)
        atomicAdd(&ksumOut[bh * 64 + tid], ksumLds[tid]);
}

// ---------------------------------------------------------------------------
// Phase 2: grid(NQ/128, BH), 256 threads (8 waves). Each wave: 16 n-rows x 64 m.
// B = KV^T fragments straight from scratch (d lane-contiguous), A = elu(Q)+1.
// ---------------------------------------------------------------------------
__global__ __launch_bounds__(256)
void la_phase2(const float* __restrict__ Qin, const float* __restrict__ kv,
               const float* __restrict__ ksum, float* __restrict__ Out)
{
    const int tid  = threadIdx.x;
    const int wave = tid >> 5;
    const int lane = tid & 31;
    const int col  = lane & 15;
    const int half = lane >> 4;
    const int bh   = blockIdx.y;
    const int b    = bh >> 3;
    const int h    = bh & 7;
    const int n0   = blockIdx.x * 128 + wave * 16;

    const float* kvb = kv   + bh * 4096;  // [m][d]
    const float* ks  = ksum + bh * 64;

    // B fragments: element i -> d = ds_*32 + half*16 + i (contiguous f32)
    Frag bf[4][2];
    #pragma unroll
    for (int mi = 0; mi < 4; ++mi)
        #pragma unroll
        for (int ds_ = 0; ds_ < 2; ++ds_) {
            const float* p = kvb + (mi * 16 + col) * 64 + ds_ * 32 + half * 16;
            #pragma unroll
            for (int i = 0; i < 16; ++i) bf[mi][ds_].e[i] = (_Float16)p[i];
        }

    // Z = 1/(Q'_row . Ksum + eps), row = n0+col
    const long qbase = (((long)b * NQ + (n0 + col)) * HQ + h) * DQ;
    const float4* qr = (const float4*)(Qin + qbase);
    const float4* kq = (const float4*)ks;
    float zacc = 0.0f;
    #pragma unroll
    for (int j = 0; j < 16; ++j) {
        float4 q = qr[j];
        float4 k = kq[j];
        zacc += elup1(q.x) * k.x + elup1(q.y) * k.y +
                elup1(q.z) * k.z + elup1(q.w) * k.w;
    }
    const float zinv = 1.0f / (zacc + EPSL);

    // A fragments: row n0+col, k(d) = ds_*32 + half*16 + i (contiguous, same as B)
    Frag af[2];
    #pragma unroll
    for (int ds_ = 0; ds_ < 2; ++ds_) {
        const float* p0 = Qin + qbase + ds_ * 32 + half * 16;
        const float* p1 = p0 + 8;
        #pragma unroll
        for (int i = 0; i < 8; ++i) {
            af[ds_].e[i]     = (_Float16)elup1(p0[i]);
            af[ds_].e[8 + i] = (_Float16)elup1(p1[i]);
        }
    }

    v8f acc[4] = {};
    #pragma unroll
    for (int mi = 0; mi < 4; ++mi)
        #pragma unroll
        for (int ds_ = 0; ds_ < 2; ++ds_)
            acc[mi] = __builtin_amdgcn_wmma_f32_16x16x32_f16(
                false, af[ds_].v, false, bf[mi][ds_].v, (short)0, acc[mi],
                false, false);

    // scale rows by Z (row owner is lane 8*half+r of this wave) and store
    #pragma unroll
    for (int r = 0; r < 8; ++r) {
        const float zr = __shfl(zinv, 8 * half + r, 32);
        const int   n  = n0 + 8 * half + r;
        const long  ob = (((long)b * NQ + n) * HQ + h) * DQ;
        #pragma unroll
        for (int mi = 0; mi < 4; ++mi)
            Out[ob + mi * 16 + col] = acc[mi][r] * zr;
    }
}

// ---------------------------------------------------------------------------
extern "C" void kernel_launch(void* const* d_in, const int* in_sizes, int n_in,
                              void* d_out, int out_size, void* d_ws, size_t ws_size,
                              hipStream_t stream) {
    const float* Q    = (const float*)d_in[0];
    const float* K    = (const float*)d_in[1];
    const float* V    = (const float*)d_in[2];
    const float* mask = (const float*)d_in[3];
    float* out  = (float*)d_out;

    float* kv   = (float*)d_ws;          // BH * 64 * 64 f32  (512 KB)
    float* ksum = kv + BH * 4096;        // BH * 64 f32       (8 KB)
    const int ztot = BH * 4096 + BH * 64;

    la_zero<<<(ztot + 255) / 256, 256, 0, stream>>>(kv, ztot);
    la_phase1<<<dim3(NQ / P1_ROWS, BH), 128, 0, stream>>>(K, V, mask, kv, ksum);
    la_phase2<<<dim3(NQ / 128, BH), 256, 0, stream>>>(Q, kv, ksum, out);
}
